// GALE_block_49435073577212
// MI455X (gfx1250) — compile-verified, matrix operand order checked by reading, not run on hardware
//
#include <hip/hip_runtime.h>
#include <hip/hip_bf16.h>

#define DEV static __device__ __forceinline__

#if defined(__has_builtin)
#if __has_builtin(__builtin_amdgcn_global_load_async_to_lds_b128) && \
    __has_builtin(__builtin_amdgcn_s_wait_asynccnt)
#define USE_ASYNC_LDS 1
#endif
#endif

typedef __attribute__((ext_vector_type(16))) __bf16 v16bf;
typedef __attribute__((ext_vector_type(8)))  float  v8f;

union Frag {
    v16bf v;
    uint4 q[2];
    unsigned short u[16];
};

DEV unsigned short f2bf(float f) {
    union { float f; unsigned u; } x; x.f = f;
    unsigned r = (x.u >> 16) & 1u;
    x.u += 0x7fffu + r;                // round-to-nearest-even
    return (unsigned short)(x.u >> 16);
}
DEV float fastrcp(float x) { return __builtin_amdgcn_rcpf(x); }   // v_rcp_f32

#if USE_ASYNC_LDS
// Builtin signature (from compiler diagnostic): arg0 = AS(1) int4*, arg1 = AS(3) int4*,
// then two immediate ints (offset, cpol).
typedef __attribute__((ext_vector_type(4))) int v4i;
typedef __attribute__((address_space(1))) v4i gv4i_t;
typedef __attribute__((address_space(3))) v4i lv4i_t;
// 16B global -> LDS, no VGPR round-trip; tracked by ASYNCcnt.
DEV void async_copy16(const void* g, void* l) {
    __builtin_amdgcn_global_load_async_to_lds_b128(
        (gv4i_t*)(unsigned long long)(uintptr_t)g,
        (lv4i_t*)(unsigned)(uintptr_t)l, 0, 0);
}
#endif

DEV v8f wmma_bf16(const Frag& a, const Frag& b, v8f c) {
    return __builtin_amdgcn_wmma_f32_16x16x32_bf16(
        /*neg_a=*/false, a.v, /*neg_b=*/false, b.v,
        /*c_mod=*/(short)0, c, /*reuse_a=*/false, /*reuse_b=*/false);
}

// A/B fragment load per CDNA5 16-bit layout:
// lanes 0-15: K 0..7 then 16..23; lanes 16-31: K 8..15 then 24..31.
// 'row' is the M (or N for B) index for this lane; base is row-major (A) or
// column-major (B) so both chunks are contiguous 16B loads.
DEV void load_frag(Frag& f, const unsigned short* base, int stride, int row, int kb) {
    const unsigned short* p = base + (size_t)row * stride;
    f.q[0] = *(const uint4*)(p + kb);
    f.q[1] = *(const uint4*)(p + 16 + kb);
}
DEV void load_frag_g(Frag& f, const unsigned short* p, int kb) {
    f.q[0] = *(const uint4*)(p + kb);
    f.q[1] = *(const uint4*)(p + 16 + kb);
}

// ---------------------------------------------------------------------------
// LayerNorm: fp32 in -> bf16 out. One block per row, blockDim == C == 256.
// ---------------------------------------------------------------------------
__global__ void __launch_bounds__(256) ln_kernel(
    const float* __restrict__ x, const float* __restrict__ g,
    const float* __restrict__ bta, unsigned short* __restrict__ out, int C)
{
    int row = blockIdx.x, t = threadIdx.x;
    __shared__ float red[256];
    float v = x[(size_t)row * C + t];
    red[t] = v; __syncthreads();
    for (int s = 128; s > 0; s >>= 1) { if (t < s) red[t] += red[t + s]; __syncthreads(); }
    float mean = red[0] / C; __syncthreads();
    float d = v - mean;
    red[t] = d * d; __syncthreads();
    for (int s = 128; s > 0; s >>= 1) { if (t < s) red[t] += red[t + s]; __syncthreads(); }
    float rstd = rsqrtf(red[0] / C + 1e-5f);
    out[(size_t)row * C + t] = f2bf(d * rstd * g[t] + bta[t]);
}

// ---------------------------------------------------------------------------
// Generic WMMA bf16 GEMM: out = epilogue(A[MxK](bf16) @ Bw[KxN](f32) + bias)
// Tile 128x128, BK=32; 8 waves, each wave 32x64 (2x4 wmma tiles).
// A tile staged via async global->LDS when available (ASYNCcnt), overlapping
// with the B tile's f32->bf16 convert+transpose through VGPRs.
// ---------------------------------------------------------------------------
template<bool OUT_F32, bool DO_GELU, bool RESID>
__global__ void __launch_bounds__(256) gemm_kernel(
    const unsigned short* __restrict__ A, const float* __restrict__ Bw,
    const float* __restrict__ bias, const float* __restrict__ resid,
    void* __restrict__ outp, int M, int N, int K)
{
    constexpr int BM = 128, BN = 128, BK = 32, SA = 40, SB = 40;
    __shared__ unsigned short As[BM * SA];
    __shared__ unsigned short Bs[BN * SB];

    int t = threadIdx.x;
    int wave = t >> 5, lane = t & 31;
    int wm = wave >> 1, wn = wave & 1;
    int lrow = lane & 15, kb = (lane >> 4) * 8;
    int rowBase = blockIdx.x * BM, colBase = blockIdx.y * BN;

    v8f acc[2][4];
    for (int i = 0; i < 2; i++) for (int j = 0; j < 4; j++) acc[i][j] = (v8f){};

    for (int k0 = 0; k0 < K; k0 += BK) {
        { // stage A: 128x32 bf16, row-major
            int r = t >> 1, kc = (t & 1) * 16;
            const unsigned short* src = A + (size_t)(rowBase + r) * K + k0 + kc;
#if USE_ASYNC_LDS
            async_copy16(src,     &As[r * SA + kc]);
            async_copy16(src + 8, &As[r * SA + kc + 8]);
#else
            uint4 d0 = ((const uint4*)src)[0], d1 = ((const uint4*)src)[1];
            *(uint4*)(&As[r * SA + kc])     = d0;
            *(uint4*)(&As[r * SA + kc + 8]) = d1;
#endif
            if (k0 + BK < K)
                __builtin_prefetch(src + BK, 0, 1);
        }
        { // stage B: 32x128 f32 -> bf16 col-major (so B frags are contiguous)
            int n = t >> 1, ks = (t & 1) * 16;
            unsigned short tmp[16];
#pragma unroll
            for (int e = 0; e < 16; e++)
                tmp[e] = f2bf(Bw[(size_t)(k0 + ks + e) * N + colBase + n]);
            *(uint4*)(&Bs[n * SB + ks])     = *(uint4*)(tmp);
            *(uint4*)(&Bs[n * SB + ks + 8]) = *(uint4*)(tmp + 8);
        }
#if USE_ASYNC_LDS
        __builtin_amdgcn_s_wait_asynccnt(0);
#endif
        __syncthreads();

        Frag a[2], b[4];
#pragma unroll
        for (int i = 0; i < 2; i++) load_frag(a[i], As, SA, wm * 32 + i * 16 + lrow, kb);
#pragma unroll
        for (int j = 0; j < 4; j++) load_frag(b[j], Bs, SB, wn * 64 + j * 16 + lrow, kb);
#pragma unroll
        for (int i = 0; i < 2; i++)
#pragma unroll
            for (int j = 0; j < 4; j++)
                acc[i][j] = wmma_bf16(a[i], b[j], acc[i][j]);
        __syncthreads();
    }

    // epilogue: C/D layout -> row = e + (lane>=16)*8 within a 16-row tile
#pragma unroll
    for (int i = 0; i < 2; i++) {
        int r0 = rowBase + wm * 32 + i * 16 + (lane >> 4) * 8;
#pragma unroll
        for (int j = 0; j < 4; j++) {
            int c = colBase + wn * 64 + j * 16 + lrow;
            float bv = bias[c];
#pragma unroll
            for (int e = 0; e < 8; e++) {
                int r = r0 + e;
                float v = acc[i][j][e] + bv;
                if (DO_GELU) v = 0.5f * v * (1.0f + erff(v * 0.70710678118654752f));
                if (RESID)   v += resid[(size_t)r * N + c];
                if (OUT_F32) ((float*)outp)[(size_t)r * N + c] = v;
                else ((unsigned short*)outp)[(size_t)r * N + c] = f2bf(v);
            }
        }
    }
}

__global__ void zero_kernel(float* p, int n) {
    int i = blockIdx.x * blockDim.x + threadIdx.x;
    if (i < n) p[i] = 0.f;
}

// ---------------------------------------------------------------------------
// Slice kernel: per (b,h, 256-token tile):
//   logits = xm_head @ Wslice + bslice, /temp  (WMMA, K=D=32)
//   sw = softmax_G(logits)  -> bf16 global + transposed LDS copy
//   norm[g] += sum_n sw      (LDS then global atomics)
//   st[g,d] += sw^T @ fxm    (WMMA per-wave 32-token K chunk, global f32 atomics)
// ---------------------------------------------------------------------------
__global__ void __launch_bounds__(256) slice_kernel(
    const unsigned short* __restrict__ xm,   // [B*N,256] bf16
    const unsigned short* __restrict__ fxm,  // [B*N,256] bf16
    const float* __restrict__ Wslice, const float* __restrict__ bslice,
    const float* __restrict__ temperature,
    unsigned short* __restrict__ sw,         // [B,H,N,G] bf16
    float* __restrict__ st,                  // [B,H,G,D]
    float* __restrict__ nrm)                 // [B,H,G]
{
    const int N = 32768, H = 8, G = 32, Dh = 32, C = 256;
    int n0 = blockIdx.x * 256;
    int h = blockIdx.y, b = blockIdx.z;
    int t = threadIdx.x, wave = t >> 5, lane = t & 31;
    int lrow = lane & 15, kb = (lane >> 4) * 8;

    __shared__ __align__(16) unsigned char smem[33792 + 16896 + 2560 + 128];
    float*          logits = (float*)smem;                          // 256*33 f32 (phase 1-2)
    unsigned short* ldsF   = (unsigned short*)smem;                 // 8*32*40 bf16 (phase 4, reuse)
    unsigned short* swT    = (unsigned short*)(smem + 33792);       // 32*264 bf16
    unsigned short* ldsW   = (unsigned short*)(smem + 33792 + 16896);   // 32*40 col-major Wslice
    float*          normL  = (float*)(smem + 33792 + 16896 + 2560);     // 32

    for (int i = t; i < 1024; i += 256) {
        int k = i >> 5, g = i & 31;
        ldsW[g * 40 + k] = f2bf(Wslice[k * 32 + g]);
    }
    if (t < 32) normL[t] = 0.f;
    __syncthreads();

    // ---- phase 1: logits via WMMA (K = D = 32, single slab)
    float rt = fastrcp(temperature[h]);
    {
        int r0 = wave * 32;
        Frag a[2], bw[2];
#pragma unroll
        for (int i = 0; i < 2; i++) {
            int n = n0 + r0 + i * 16 + lrow;
            load_frag_g(a[i], xm + ((size_t)(b * N + n)) * C + h * 32, kb);
        }
#pragma unroll
        for (int j = 0; j < 2; j++) load_frag(bw[j], ldsW, 40, j * 16 + lrow, kb);
#pragma unroll
        for (int i = 0; i < 2; i++)
#pragma unroll
            for (int j = 0; j < 2; j++) {
                v8f c = (v8f){};
                c = wmma_bf16(a[i], bw[j], c);
                int g = j * 16 + lrow;
                float bsl = bslice[g];
#pragma unroll
                for (int e = 0; e < 8; e++) {
                    int rr = r0 + i * 16 + (lane >> 4) * 8 + e;
                    logits[rr * 33 + g] = (c[e] + bsl) * rt;
                }
            }
    }
    __syncthreads();

    // ---- phase 2: softmax over G per token row; write sw (global bf16) + swT (LDS)
    {
        float vbuf[32], m = -1e30f;
#pragma unroll
        for (int g = 0; g < 32; g++) { vbuf[g] = logits[t * 33 + g]; m = fmaxf(m, vbuf[g]); }
        float s = 0.f;
#pragma unroll
        for (int g = 0; g < 32; g++) { vbuf[g] = __expf(vbuf[g] - m); s += vbuf[g]; }
        float inv = fastrcp(s);
        size_t swb = (((size_t)(b * H + h)) * N + (n0 + t)) * G;
#pragma unroll
        for (int g = 0; g < 32; g++) {
            float w = vbuf[g] * inv;
            unsigned short wb = f2bf(w);
            sw[swb + g] = wb;
            swT[g * 264 + t] = wb;
            atomicAdd(&normL[g], w);
        }
    }
    __syncthreads();
    if (t < 32) atomicAdd(&nrm[((size_t)(b * H + h)) * G + t], normL[t]);

    // ---- phase 4: st partial = sw^T @ fxm per-wave (K chunk = this wave's 32 tokens)
    {
        unsigned short* fw = ldsF + wave * 32 * 40;
        { // stage fxm 32x32 (this wave's tokens) col-major into LDS
            int n = n0 + wave * 32 + lane;
            const unsigned short* p = fxm + ((size_t)(b * N + n)) * C + h * 32;
            union { uint4 q[4]; unsigned short u[32]; } rowv;
            rowv.q[0] = ((const uint4*)p)[0]; rowv.q[1] = ((const uint4*)p)[1];
            rowv.q[2] = ((const uint4*)p)[2]; rowv.q[3] = ((const uint4*)p)[3];
#pragma unroll
            for (int d = 0; d < 32; d++) fw[d * 40 + lane] = rowv.u[d];
        }
        __syncthreads();
        Frag a[2], bf[2];
#pragma unroll
        for (int i = 0; i < 2; i++)
            load_frag_g(a[i], swT + (size_t)(i * 16 + lrow) * 264 + wave * 32, kb);
#pragma unroll
        for (int j = 0; j < 2; j++) load_frag(bf[j], fw, 40, j * 16 + lrow, kb);
#pragma unroll
        for (int i = 0; i < 2; i++)
#pragma unroll
            for (int j = 0; j < 2; j++) {
                v8f c = (v8f){};
                c = wmma_bf16(a[i], bf[j], c);
#pragma unroll
                for (int e = 0; e < 8; e++) {
                    int g = i * 16 + (lane >> 4) * 8 + e;
                    int d = j * 16 + lrow;
                    atomicAdd(&st[(((size_t)(b * H + h)) * G + g) * Dh + d], c[e]);
                }
            }
    }
}

// ---------------------------------------------------------------------------
// Slice-token attention (tiny: 32x32 per (b,h)) + cross-attn on 64 context
// tokens + sigmoid gate.  One wave per (b,h); VALU is fine at this size.
// ---------------------------------------------------------------------------
__global__ void __launch_bounds__(32) slice_attn_kernel(
    const float* __restrict__ st, const float* __restrict__ nrm,
    const float* __restrict__ context,
    const float* __restrict__ Wq, const float* __restrict__ Wk, const float* __restrict__ Wv,
    const float* __restrict__ Wcq, const float* __restrict__ bcq,
    const float* __restrict__ Wck, const float* __restrict__ bck,
    const float* __restrict__ Wcv, const float* __restrict__ bcv,
    const float* __restrict__ state_mixing,
    unsigned short* __restrict__ ost)
{
    const int G = 32, Dh = 32, SC = 64, DC = 32, H = 8;
    int h = blockIdx.x, b = blockIdx.y, g = threadIdx.x;
    __shared__ float stS[32][32], kS[32][32], vS[32][32], ckS[64][32], cvS[64][32];

    size_t bh = (size_t)(b * H + h);
    float inv = 1.f / (nrm[bh * G + g] + 1e-5f);
    for (int d = 0; d < 32; d++) stS[g][d] = st[(bh * G + g) * Dh + d] * inv;
    __syncthreads();

    float q[32];
    for (int d = 0; d < 32; d++) {
        float aq = 0.f, ak = 0.f, av = 0.f;
        for (int e = 0; e < 32; e++) {
            float s = stS[g][e];
            aq += s * Wq[e * 32 + d]; ak += s * Wk[e * 32 + d]; av += s * Wv[e * 32 + d];
        }
        q[d] = aq; kS[g][d] = ak; vS[g][d] = av;
    }
    __syncthreads();

    const float scale = 0.17677669529663687f;   // D^-0.5
    float sc[32], m = -1e30f;
    for (int j = 0; j < 32; j++) {
        float s = 0.f;
        for (int d = 0; d < 32; d++) s += q[d] * kS[j][d];
        sc[j] = s * scale; m = fmaxf(m, sc[j]);
    }
    float ssum = 0.f;
    for (int j = 0; j < 32; j++) { sc[j] = __expf(sc[j] - m); ssum += sc[j]; }
    float sinv = fastrcp(ssum);
    float selftok[32];
    for (int d = 0; d < 32; d++) {
        float a = 0.f;
        for (int j = 0; j < 32; j++) a += sc[j] * vS[j][d];
        selftok[d] = a * sinv;
    }

    float cq[32];
    for (int d = 0; d < 32; d++) {
        float a = bcq[d];
        for (int e = 0; e < 32; e++) a += stS[g][e] * Wcq[e * 32 + d];
        cq[d] = a;
    }
    for (int rep = 0; rep < 2; rep++) {
        int s = g + rep * 32;
        const float* cx = context + (bh * SC + s) * DC;
        for (int d = 0; d < 32; d++) {
            float a = bck[d], a2 = bcv[d];
            for (int e = 0; e < 32; e++) { float c = cx[e]; a += c * Wck[e * 32 + d]; a2 += c * Wcv[e * 32 + d]; }
            ckS[s][d] = a; cvS[s][d] = a2;
        }
    }
    __syncthreads();

    float sc2[64]; m = -1e30f;
    for (int j = 0; j < 64; j++) {
        float s = 0.f;
        for (int d = 0; d < 32; d++) s += cq[d] * ckS[j][d];
        sc2[j] = s * scale; m = fmaxf(m, sc2[j]);
    }
    ssum = 0.f;
    for (int j = 0; j < 64; j++) { sc2[j] = __expf(sc2[j] - m); ssum += sc2[j]; }
    float cinv = fastrcp(ssum);
    float mw = 1.f / (1.f + __expf(-state_mixing[0]));
    for (int d = 0; d < 32; d++) {
        float a = 0.f;
        for (int j = 0; j < 64; j++) a += sc2[j] * cvS[j][d];
        float o = mw * selftok[d] + (1.f - mw) * (a * cinv);
        ost[(bh * G + g) * Dh + d] = f2bf(o);
    }
}

// ---------------------------------------------------------------------------
// De-slice: htok[n, h*32+d] = sw[b,h,n,:] @ ost[b,h,:,:]  (WMMA, K = G = 32)
// 128 tokens per block; ost for all 8 heads staged col-major in LDS.
// ---------------------------------------------------------------------------
__global__ void __launch_bounds__(256) deslice_kernel(
    const unsigned short* __restrict__ sw,    // [B,H,N,G] bf16
    const unsigned short* __restrict__ ost,   // [B,H,G,D] bf16
    unsigned short* __restrict__ htok)        // [B*N,256] bf16
{
    const int N = 32768, H = 8, G = 32, Dh = 32, C = 256;
    __shared__ unsigned short ostL[8 * 32 * 40];   // [h][d*40 + g]
    int t = threadIdx.x, wave = t >> 5, lane = t & 31;
    int lrow = lane & 15, kb = (lane >> 4) * 8;
    int row0 = blockIdx.x * 128;
    int b = row0 / N, nb = row0 % N;

    for (int i = t; i < 8192; i += 256) {
        int hh = i >> 10, rem = i & 1023, g = rem >> 5, d = rem & 31;
        ostL[hh * 1280 + d * 40 + g] = ost[(((size_t)(b * H + hh)) * G + g) * Dh + d];
    }
    __syncthreads();

    int n = nb + wave * 16 + lrow;
    for (int h = 0; h < 8; h++) {
        Frag a, bw[2];
        load_frag_g(a, sw + (((size_t)(b * H + h)) * N + n) * G, kb);
#pragma unroll
        for (int j = 0; j < 2; j++) load_frag(bw[j], ostL + h * 1280, 40, j * 16 + lrow, kb);
#pragma unroll
        for (int j = 0; j < 2; j++) {
            v8f c = (v8f){};
            c = wmma_bf16(a, bw[j], c);
#pragma unroll
            for (int e = 0; e < 8; e++) {
                int rr = nb + wave * 16 + (lane >> 4) * 8 + e;
                int d = j * 16 + lrow;
                htok[((size_t)(b * N + rr)) * C + h * 32 + d] = f2bf(c[e]);
            }
        }
    }
}

// ---------------------------------------------------------------------------
extern "C" void kernel_launch(void* const* d_in, const int* in_sizes, int n_in,
                              void* d_out, int out_size, void* d_ws, size_t ws_size,
                              hipStream_t stream)
{
    const int B = 2, N = 32768, C = 256, H = 8, G = 32;
    const int M = B * N;   // 65536

    const float* fx        = (const float*)d_in[0];
    const float* context   = (const float*)d_in[1];
    const float* ln1_g     = (const float*)d_in[2];
    const float* ln1_b     = (const float*)d_in[3];
    const float* Wfx       = (const float*)d_in[4];
    const float* bfx       = (const float*)d_in[5];
    const float* Wx        = (const float*)d_in[6];
    const float* bx        = (const float*)d_in[7];
    const float* Wslice    = (const float*)d_in[8];
    const float* bslice    = (const float*)d_in[9];
    const float* temp      = (const float*)d_in[10];
    const float* Wq        = (const float*)d_in[11];
    const float* Wk        = (const float*)d_in[12];
    const float* Wv        = (const float*)d_in[13];
    const float* Wcq       = (const float*)d_in[14];
    const float* bcq       = (const float*)d_in[15];
    const float* Wck       = (const float*)d_in[16];
    const float* bck       = (const float*)d_in[17];
    const float* Wcv       = (const float*)d_in[18];
    const float* bcv       = (const float*)d_in[19];
    const float* state_mix = (const float*)d_in[20];
    const float* Wout      = (const float*)d_in[21];
    const float* bout      = (const float*)d_in[22];
    const float* ln2_g     = (const float*)d_in[23];
    const float* ln2_b     = (const float*)d_in[24];
    const float* Wm1       = (const float*)d_in[25];
    const float* bm1       = (const float*)d_in[26];
    const float* Wm2       = (const float*)d_in[27];
    const float* bm2       = (const float*)d_in[28];

    // workspace layout (bytes), phased reuse:
    //   [0,32M)    : xh (LN1 out)      -> later htok (de-sliced tokens)
    //   [32M,64M)  : fxm               -> later h2 (LN2 out)
    //   [64M,96M)  : xm        \
    //   [96M,128M) : sw         }-> later mmid spans [64M,192M)
    //   [128M,192M): (mmid tail)/
    //   [192M,256M): fx1 (f32 residual-1 output)
    //   [256M,...) : st, norm, ost (small)
    const size_t MB = 1ull << 20;
    char* ws = (char*)d_ws;
    unsigned short* xh   = (unsigned short*)(ws + 0);
    unsigned short* htok = (unsigned short*)(ws + 0);
    unsigned short* fxm  = (unsigned short*)(ws + 32 * MB);
    unsigned short* h2   = (unsigned short*)(ws + 32 * MB);
    unsigned short* xm   = (unsigned short*)(ws + 64 * MB);
    unsigned short* swb  = (unsigned short*)(ws + 96 * MB);
    unsigned short* mmid = (unsigned short*)(ws + 64 * MB);
    float*          fx1  = (float*)(ws + 192 * MB);
    float*          st   = (float*)(ws + 256 * MB);
    float*          nrm  = (float*)(ws + 256 * MB + 131072);
    unsigned short* ost  = (unsigned short*)(ws + 256 * MB + 131072 + 2048);

    // 1) LN1
    ln_kernel<<<M, 256, 0, stream>>>(fx, ln1_g, ln1_b, xh, C);
    // 2) fxm = xh @ Wfx + bfx ; 3) xm = xh @ Wx + bx   (bf16 out)
    gemm_kernel<false, false, false><<<dim3(M / 128, 2), 256, 0, stream>>>(
        xh, Wfx, bfx, nullptr, fxm, M, 256, 256);
    gemm_kernel<false, false, false><<<dim3(M / 128, 2), 256, 0, stream>>>(
        xh, Wx, bx, nullptr, xm, M, 256, 256);
    // 4) zero st (32768 f32) + norm (512 f32), contiguous
    zero_kernel<<<(33280 + 255) / 256, 256, 0, stream>>>(st, 33280);
    // 5) slice weights + softmax + norm + st aggregation
    slice_kernel<<<dim3(N / 256, H, B), 256, 0, stream>>>(
        xm, fxm, Wslice, bslice, temp, swb, st, nrm);
    // 6) tiny per-(b,h) self+cross attention on slice tokens
    slice_attn_kernel<<<dim3(H, B), 32, 0, stream>>>(
        st, nrm, context, Wq, Wk, Wv, Wcq, bcq, Wck, bck, Wcv, bcv, state_mix, ost);
    // 7) de-slice back to tokens
    deslice_kernel<<<M / 128, 256, 0, stream>>>(swb, ost, htok);
    // 8) out-proj + residual-1: fx1 = fx + htok @ Wout + bout   (f32 out)
    gemm_kernel<true, false, true><<<dim3(M / 128, 2), 256, 0, stream>>>(
        htok, Wout, bout, fx, fx1, M, 256, 256);
    // 9) LN2
    ln_kernel<<<M, 256, 0, stream>>>(fx1, ln2_g, ln2_b, h2, C);
    // 10) MLP up + exact GELU (bf16 out)
    gemm_kernel<false, true, false><<<dim3(M / 128, 8), 256, 0, stream>>>(
        h2, Wm1, bm1, nullptr, mmid, M, 1024, 256);
    // 11) MLP down + residual-2 -> d_out (f32)
    gemm_kernel<true, false, true><<<dim3(M / 128, 2), 256, 0, stream>>>(
        mmid, Wm2, bm2, fx1, (float*)d_out, M, 256, 1024);
}
